// nwlm_61280593380185
// MI455X (gfx1250) — matrix operand order repaired and checked by reading, not compiled, not run on hardware
//
#include <hip/hip_runtime.h>
#include <hip/hip_bf16.h>

typedef __attribute__((ext_vector_type(16))) __bf16 v16bf;
typedef __attribute__((ext_vector_type(8)))  float  v8f;

#define EPSF 1e-12f

union AF { v16bf v; unsigned int u[8]; };

__device__ __forceinline__ v8f wmma_bf16(const AF& a, const AF& b, v8f c) {
  // D = A(16x32 bf16) * B(32x16 bf16) + C(16x16 f32)
  return __builtin_amdgcn_wmma_f32_16x16x32_bf16(false, a.v, false, b.v,
                                                 (short)0, c, false, false);
}

// CDNA5 async global->LDS copy (tracked with ASYNCcnt). Inline asm per
// CDNA5 ISA; portable across ROCm7.2 / amdgpu-toolchain.
__device__ __forceinline__ void async_load_b128(unsigned lds_addr,
                                                unsigned long long gaddr) {
  asm volatile("global_load_async_to_lds_b128 %0, %1, off"
               :: "v"(lds_addr), "v"(gaddr) : "memory");
}
__device__ __forceinline__ void wait_asynccnt0() {
  asm volatile("s_wait_asynccnt 0x0" ::: "memory");
}

// ---------------- prep kernels ----------------

// z (4096 rows x 8 heads x 64) -> bf16 copy + per-(row,head) sum of squares
__global__ void prep_z(const float* __restrict__ z, __bf16* __restrict__ zb,
                       float* __restrict__ z2) {
  const int rid = blockIdx.x;        // 0 .. 4096*8-1  (= row*8 + head)
  const int t = threadIdx.x;         // 0..31
  const float a = z[(size_t)rid * 64 + t];
  const float b = z[(size_t)rid * 64 + t + 32];
  zb[(size_t)rid * 64 + t]      = (__bf16)a;
  zb[(size_t)rid * 64 + t + 32] = (__bf16)b;
  float ss = a * a + b * b;
#pragma unroll
  for (int o = 16; o > 0; o >>= 1) ss += __shfl_xor(ss, o, 32);
  if (t == 0) z2[rid] = ss;
}

// Landmarks: R -> bf16, cs = {R2, (relu(q)+eps)^-0.5}, W -> bf16 transposed (h,dh,N)
__global__ void prep_lm(const float* __restrict__ R, const float* __restrict__ q,
                        const float* __restrict__ W, __bf16* __restrict__ Rb,
                        float2* __restrict__ cs, __bf16* __restrict__ Wbt) {
  const int hn = blockIdx.x;         // 0 .. 8*2048-1
  const int t = threadIdx.x;         // 0..31
  const int head = hn >> 11, n = hn & 2047;
  const float a = R[(size_t)hn * 64 + t];
  const float b = R[(size_t)hn * 64 + t + 32];
  Rb[(size_t)hn * 64 + t]      = (__bf16)a;
  Rb[(size_t)hn * 64 + t + 32] = (__bf16)b;
  float ss = a * a + b * b;
#pragma unroll
  for (int o = 16; o > 0; o >>= 1) ss += __shfl_xor(ss, o, 32);
  if (t == 0)
    cs[hn] = make_float2(ss, rsqrtf(fmaxf(q[hn], 0.0f) + EPSF)); // alpha = 0.5
  const float w0 = W[(size_t)hn * 64 + t];
  const float w1 = W[(size_t)hn * 64 + t + 32];
  Wbt[(size_t)head * 131072 + (size_t)t * 2048 + n]        = (__bf16)w0;
  Wbt[(size_t)head * 131072 + (size_t)(t + 32) * 2048 + n] = (__bf16)w1;
}

// W_O (512x512) -> bf16 transposed (nout, k)
__global__ void prep_wo(const float* __restrict__ WO, __bf16* __restrict__ WOt) {
  const int idx = blockIdx.x * 256 + threadIdx.x;  // 0..262143 = k*512 + n
  const int k = idx >> 9, n = idx & 511;
  WOt[(size_t)n * 512 + k] = (__bf16)WO[idx];
}

// ---------------- main fused kernel ----------------
// grid = (8 heads, 32 row-tiles of 128), block = 256 (8 waves; each wave: 16 rows)
__global__ __launch_bounds__(256)
void nwlm_main(const __bf16* __restrict__ zb, const float* __restrict__ z2,
               const __bf16* __restrict__ Rb, const float2* __restrict__ cs,
               const __bf16* __restrict__ Wbt, __bf16* __restrict__ Yb) {
  // double-buffered chunk tiles filled by async global->LDS DMA
  __shared__ __align__(16) unsigned int ldsRbuf[2][1024]; // 32 lm x 64 bf16 each
  __shared__ __align__(16) unsigned int ldsWbuf[2][1024]; // 64 dh x 32 bf16 each
  __shared__ __align__(16) unsigned int ldsKq[8][256];    // per-wave 16x32 bf16

  const int head = blockIdx.x;
  const int tid = threadIdx.x;
  const int wave = tid >> 5;
  const int lane = tid & 31;
  const int lm = lane & 15;
  const int hi = lane >> 4;
  const int r0 = blockIdx.y * 128 + wave * 16;

  const unsigned int* RbU = (const unsigned int*)(Rb + (size_t)head * 2048 * 64);
  const unsigned int* WtU = (const unsigned int*)(Wbt + (size_t)head * 64 * 2048);
  const float2* csh = cs + head * 2048;

  const unsigned ldsRbase0 = (unsigned)(uintptr_t)&ldsRbuf[0][0];
  const unsigned ldsRbase1 = (unsigned)(uintptr_t)&ldsRbuf[1][0];
  const unsigned ldsWbase0 = (unsigned)(uintptr_t)&ldsWbuf[0][0];
  const unsigned ldsWbase1 = (unsigned)(uintptr_t)&ldsWbuf[1][0];
  const int dh = tid >> 2, q4 = tid & 3;

  auto issue_chunk = [&](int nc, int buf) {
    const unsigned rb = (buf ? ldsRbase1 : ldsRbase0) + (unsigned)tid * 16u;
    const unsigned wb = (buf ? ldsWbase1 : ldsWbase0) + (unsigned)tid * 16u;
    async_load_b128(rb, (unsigned long long)(uintptr_t)(RbU + nc * 32 + tid * 4));
    async_load_b128(wb, (unsigned long long)(uintptr_t)(WtU + dh * 1024 + nc / 2 + q4 * 4));
  };

  // A-fragments of z (16 rows x 64 features): ISA 16-bit A 16x32 layout
  AF az0, az1;
  {
    const unsigned int* zu = (const unsigned int*)zb;
    const unsigned int base = (unsigned)((r0 + lm) * 8 + head) * 32u;
#pragma unroll
    for (int j = 0; j < 4; ++j) {
      az0.u[j]     = zu[base + hi * 4 + j];          // K = hi*8 + 2j
      az0.u[4 + j] = zu[base + 8 + hi * 4 + j];      // K = 16 + hi*8 + 2j
      az1.u[j]     = zu[base + 16 + hi * 4 + j];
      az1.u[4 + j] = zu[base + 24 + hi * 4 + j];
    }
  }
  // z^2 for the 8 rows this lane owns in the C/D layout (row = v + 8*hi)
  float z2r[8];
#pragma unroll
  for (int v = 0; v < 8; ++v) z2r[v] = z2[(r0 + v + 8 * hi) * 8 + head];

  AF ones;
#pragma unroll
  for (int i = 0; i < 16; ++i) ones.v[i] = (__bf16)1.0f;

  v8f accY[4] = {};
  v8f accR = {};

  issue_chunk(0, 0);  // prologue: fill buffer 0

  for (int nc = 0; nc < 2048; nc += 32) {
    const int buf = (nc >> 5) & 1;
    wait_asynccnt0();   // my async loads for this buffer have landed
    __syncthreads();    // everyone's landed; previous compute fully drained
    if (nc + 32 < 2048) issue_chunk(nc + 32, buf ^ 1);

    const unsigned int* ldsR  = ldsRbuf[buf];
    const unsigned int* ldsWt = ldsWbuf[buf];

    // ---- GEMM1: S = z @ R^T  (16 x 32, two 16x16 tiles, K=64) ----
#pragma unroll
    for (int ntile = 0; ntile < 2; ++ntile) {
      AF br0, br1;
      const int nrow = ntile * 16 + lm;
#pragma unroll
      for (int j = 0; j < 8; ++j) {
        br0.u[j] = ldsR[nrow * 32 + (j >> 2) * 8 + hi * 4 + (j & 3)];        // K = (j>>2)*16 + hi*8 + 2(j&3)
        br1.u[j] = ldsR[nrow * 32 + 16 + (j >> 2) * 8 + hi * 4 + (j & 3)];   // K = 32 + (j>>2)*16 + hi*8 + 2(j&3)
      }
      v8f s = {};
      s = wmma_bf16(az0, br0, s);
      s = wmma_bf16(az1, br1, s);
      // elementwise: dist2 -> exp -> * q^-alpha; stash bf16 tile for A-relayout
      const float2 c2 = csh[nc + ntile * 16 + lm];      // {R2, q^-alpha}
      __bf16* kqp = (__bf16*)ldsKq[wave];
#pragma unroll
      for (int v = 0; v < 8; ++v) {
        float d2 = z2r[v] + c2.x - 2.0f * s[v];
        d2 = fmaxf(d2, 0.0f);
        const float kq = __expf(-0.5f * d2) * c2.y;     // BETA = 0.5
        kqp[(v + 8 * hi) * 32 + ntile * 16 + lm] = (__bf16)kq;
      }
    }
    // intra-wave LDS RAW on the Kq staging tile
    asm volatile("s_wait_dscnt 0" ::: "memory");

    // reload Kq tile in A-matrix layout
    AF ak;
    {
      const unsigned int* kqu = ldsKq[wave];
#pragma unroll
      for (int j = 0; j < 4; ++j) {
        ak.u[j]     = kqu[lm * 16 + hi * 4 + j];
        ak.u[4 + j] = kqu[lm * 16 + 8 + hi * 4 + j];
      }
    }
    // row-sum accumulator via ones-matrix WMMA (all 16 cols hold rsum)
    accR = wmma_bf16(ak, ones, accR);

    // ---- GEMM2: Y += Kq @ W  (16 x 64) ----
#pragma unroll
    for (int dtile = 0; dtile < 4; ++dtile) {
      AF bw;
      const int dhl = dtile * 16 + lm;
#pragma unroll
      for (int j = 0; j < 8; ++j) bw.u[j] = ldsWt[dhl * 16 + (j >> 2) * 8 + hi * 4 + (j & 3)];
      accY[dtile] = wmma_bf16(ak, bw, accY[dtile]);
    }
  }

  // finalize: normalize by row sums, store bf16 Y tile
  float rinv[8];
#pragma unroll
  for (int v = 0; v < 8; ++v) rinv[v] = 1.0f / fmaxf(accR[v], EPSF);
#pragma unroll
  for (int dtile = 0; dtile < 4; ++dtile) {
    const int col = head * 64 + dtile * 16 + lm;
#pragma unroll
    for (int v = 0; v < 8; ++v) {
      const int row = r0 + v + 8 * hi;
      Yb[(size_t)row * 512 + col] = (__bf16)(accY[dtile][v] * rinv[v]);
    }
  }
}

// ---------------- output GEMM: out = Y @ W_O + b_O ----------------
// grid = 256 (16-row tiles), block = 256 (8 waves; wave w covers cols w*64..w*64+63)
__global__ __launch_bounds__(256)
void nwlm_out(const __bf16* __restrict__ Yb, const __bf16* __restrict__ WOt,
              const float* __restrict__ bO, float* __restrict__ out) {
  const int tid = threadIdx.x;
  const int wave = tid >> 5;
  const int lane = tid & 31;
  const int lm = lane & 15, hi = lane >> 4;
  const int r0 = blockIdx.x * 16;
  const int cbase = wave * 64;

  const unsigned int* YU = (const unsigned int*)Yb;
  const unsigned int* WU = (const unsigned int*)WOt;

  v8f acc[4] = {};
  for (int kc = 0; kc < 512; kc += 32) {
    AF a;
    const unsigned int abase = (unsigned)(r0 + lm) * 256 + (kc >> 1);
#pragma unroll
    for (int j = 0; j < 4; ++j) {
      a.u[j]     = YU[abase + hi * 4 + j];
      a.u[4 + j] = YU[abase + 8 + hi * 4 + j];
    }
#pragma unroll
    for (int dtile = 0; dtile < 4; ++dtile) {
      AF b;
      const unsigned int bbase =
          (unsigned)(cbase + dtile * 16 + lm) * 256 + (kc >> 1);
#pragma unroll
      for (int j = 0; j < 8; ++j) b.u[j] = WU[bbase + (j >> 2) * 8 + hi * 4 + (j & 3)];
      acc[dtile] = wmma_bf16(a, b, acc[dtile]);
    }
  }
#pragma unroll
  for (int dtile = 0; dtile < 4; ++dtile) {
    const int n = cbase + dtile * 16 + lm;
    const float bias = bO[n];
#pragma unroll
    for (int v = 0; v < 8; ++v)
      out[(size_t)(r0 + v + 8 * hi) * 512 + n] = acc[dtile][v] + bias;
  }
}

// ---------------- host launcher ----------------
extern "C" void kernel_launch(void* const* d_in, const int* in_sizes, int n_in,
                              void* d_out, int out_size, void* d_ws, size_t ws_size,
                              hipStream_t stream) {
  const float* z  = (const float*)d_in[0];   // (4,1024,8,64)
  const float* R  = (const float*)d_in[1];   // (8,2048,64)
  const float* q  = (const float*)d_in[2];   // (8,2048)
  const float* W  = (const float*)d_in[3];   // (8,2048,64)
  const float* WO = (const float*)d_in[4];   // (512,512)
  const float* bO = (const float*)d_in[5];   // (512)
  float* out = (float*)d_out;                // (4,1024,512) f32

  char* w = (char*)d_ws;
  __bf16* zb  = (__bf16*)w; w += (size_t)4096 * 8 * 64 * 2;   // 4 MB
  float*  z2  = (float*)w;  w += (size_t)4096 * 8 * 4;        // 128 KB
  __bf16* Rb  = (__bf16*)w; w += (size_t)8 * 2048 * 64 * 2;   // 2 MB
  float2* cs  = (float2*)w; w += (size_t)8 * 2048 * 8;        // 128 KB
  __bf16* Wbt = (__bf16*)w; w += (size_t)8 * 2048 * 64 * 2;   // 2 MB
  __bf16* WOt = (__bf16*)w; w += (size_t)512 * 512 * 2;       // 512 KB
  __bf16* Yb  = (__bf16*)w; w += (size_t)4096 * 512 * 2;      // 4 MB

  prep_z<<<4096 * 8, 32, 0, stream>>>(z, zb, z2);
  prep_lm<<<8 * 2048, 32, 0, stream>>>(R, q, W, Rb, cs, Wbt);
  prep_wo<<<1024, 256, 0, stream>>>(WO, WOt);

  dim3 gA(8, 32);
  nwlm_main<<<gA, 256, 0, stream>>>(zb, z2, Rb, cs, Wbt, Yb);
  nwlm_out<<<256, 256, 0, stream>>>(Yb, WOt, bO, out);
}
